// Attn_Orca_15144054686378
// MI455X (gfx1250) — compile-verified, matrix operand order checked by reading, not run on hardware
//
#include <hip/hip_runtime.h>
#include <hip/hip_bf16.h>

// CDNA5 / gfx1250 flash-attention, bf16 WMMA with f32 accumulate.
// B=2, S=2048, H=16, D=128. Wave32; 8 waves/block; 16 q-rows per wave.
// KT=64 keys/tile, double-buffered LDS, register-staged global pipeline,
// v_permlane16 VALU butterflies, lane-partial denominators, log2-domain
// softmax (scale and log2(e) folded into Q), uniform-branch rescale skip.

typedef __attribute__((ext_vector_type(16))) __bf16 v16bf;
typedef __attribute__((ext_vector_type(8)))  float  v8f;

#define B_     2
#define S_     2048
#define H_     16
#define D_     128
#define NE_    (H_ * D_)      // 2048
#define KT_    64             // keys per tile
#define NT_    (S_ / KT_)     // 32 key tiles
#define QW_    16             // q rows per wave
#define WAVES_ 8
#define QB_    (QW_ * WAVES_) // 128 q rows per block

// 16-bit WMMA A-layout / transposed-B-layout element offsets:
// lane half 'hi' selects K subset {0..7,16..23} vs {8..15,24..31}.
__device__ __forceinline__ int off_base(int v, int hi) {
    int o = (v < 4) ? (2 * v) : (2 * v + 8);   // 0,2,4,6,16,18,20,22
    return o + (hi << 3);
}

__device__ __forceinline__ v16bf load_row_frag(const __bf16* row, int hi) {
    v16bf r;
#pragma unroll
    for (int v = 0; v < 8; ++v) {
        int o = off_base(v, hi);
        r[2 * v]     = row[o];
        r[2 * v + 1] = row[o + 1];
    }
    return r;
}

// xor-butterfly permutations within 16-lane rows, on the VALU crossbar.
#define PL16(v, s0, s1) \
    __builtin_amdgcn_permlane16((v), (v), (s0), (s1), false, false)

__device__ __forceinline__ float xmax16(float x) {
    int v = __float_as_int(x); int t;
    t = PL16(v, 0x67452301u, 0xEFCDAB89u);              // xor 1
    x = fmaxf(x, __int_as_float(t)); v = __float_as_int(x);
    t = PL16(v, 0x54761032u, 0xDCFE98BAu);              // xor 2
    x = fmaxf(x, __int_as_float(t)); v = __float_as_int(x);
    t = PL16(v, 0x32107654u, 0xBA98FEDCu);              // xor 4
    x = fmaxf(x, __int_as_float(t)); v = __float_as_int(x);
    t = PL16(v, 0xFEDCBA98u, 0x76543210u);              // xor 8
    x = fmaxf(x, __int_as_float(t));
    return x;
}

__device__ __forceinline__ float xsum16(float x) {
    int v = __float_as_int(x); int t;
    t = PL16(v, 0x67452301u, 0xEFCDAB89u);
    x = x + __int_as_float(t); v = __float_as_int(x);
    t = PL16(v, 0x54761032u, 0xDCFE98BAu);
    x = x + __int_as_float(t); v = __float_as_int(x);
    t = PL16(v, 0x32107654u, 0xBA98FEDCu);
    x = x + __int_as_float(t); v = __float_as_int(x);
    t = PL16(v, 0xFEDCBA98u, 0x76543210u);
    x = x + __int_as_float(t);
    return x;
}

__global__ __launch_bounds__(256, 1)
void attn_fa_wmma(const float* __restrict__ q,
                  const float* __restrict__ k,
                  const float* __restrict__ v,
                  float* __restrict__ out)
{
    __shared__ __bf16 lK [2][KT_][D_ + 8];        // K tiles, row-major
    __shared__ __bf16 lVT[2][D_][KT_ + 8];        // V tiles, [d][perm(k)]
    __shared__ __bf16 lP [WAVES_][16][KT_ + 8];   // per-wave P, perm(k) cols

    const int tid  = threadIdx.x;
    const int lane = tid & 31;
    const int w    = tid >> 5;
    const int hi   = (lane >> 4) & 1;
    const int ln   = lane & 15;

    const int b  = blockIdx.z;
    const int h  = blockIdx.y;
    const int q0 = blockIdx.x * QB_ + w * QW_;

    // ---- Q tile -> 4 bf16 A-fragments ----
    // Fold BOTH the 1/D score scale and log2(e) into Q, so scores are in
    // the log2 domain and every exponential is a single v_exp_f32 (exp2).
    const float  qscale = 1.44269504088896340736f / (float)D_;
    const float* qrow = q + ((size_t)b * S_ + (size_t)(q0 + ln)) * NE_ + h * D_;
    v16bf qf[4];
#pragma unroll
    for (int c = 0; c < 4; ++c) {
#pragma unroll
        for (int vv = 0; vv < 8; ++vv) {
            int o = c * 32 + off_base(vv, hi);
            float2 f2 = *(const float2*)(qrow + o);
            qf[c][2 * vv]     = (__bf16)(f2.x * qscale);
            qf[c][2 * vv + 1] = (__bf16)(f2.y * qscale);
        }
    }

    // ---- accumulators ----
    v8f o_acc[8];
#pragma unroll
    for (int dt = 0; dt < 8; ++dt)
#pragma unroll
        for (int j = 0; j < 8; ++j) o_acc[dt][j] = 0.0f;

    float m_r[8], l_part[8];   // running max (log2 domain), lane-partial sums
#pragma unroll
    for (int j = 0; j < 8; ++j) { m_r[j] = -3.0e38f; l_part[j] = 0.0f; }

    // Fill mapping: thread -> (key row 0..63, 32-wide d slab)
    const int fr = tid >> 2;          // 0..63
    const int fc = (tid & 3) * 32;    // 0,32,64,96
    // Key permutation applied identically to P columns and V rows:
    // pi(16*t + ln) = 4*ln + t  -> at commit, key fr lands in column:
    const int frp = 4 * (fr & 15) + (fr >> 4);

    // ---- register staging for the global->LDS software pipeline ----
    float4 sk[8], sv[8];
    {
        const size_t base = ((size_t)b * S_ + (size_t)fr) * NE_ + h * D_ + fc;
        const float4* kp = (const float4*)(k + base);
        const float4* vp = (const float4*)(v + base);
#pragma unroll
        for (int i = 0; i < 8; ++i) { sk[i] = kp[i]; sv[i] = vp[i]; }
    }

    for (int kt = 0; kt < NT_; ++kt) {
        const int buf = kt & 1;

        // ---- commit staged tile (fp32 -> bf16) ----
#pragma unroll
        for (int i = 0; i < 8; ++i) {
            int d = fc + i * 4;
            lK[buf][fr][d + 0] = (__bf16)sk[i].x;
            lK[buf][fr][d + 1] = (__bf16)sk[i].y;
            lK[buf][fr][d + 2] = (__bf16)sk[i].z;
            lK[buf][fr][d + 3] = (__bf16)sk[i].w;
            lVT[buf][d + 0][frp] = (__bf16)sv[i].x;
            lVT[buf][d + 1][frp] = (__bf16)sv[i].y;
            lVT[buf][d + 2][frp] = (__bf16)sv[i].z;
            lVT[buf][d + 3][frp] = (__bf16)sv[i].w;
        }
        __syncthreads();

        // ---- issue next tile's global loads (consumed next iteration) ----
        if (kt + 1 < NT_) {
            const size_t base = ((size_t)b * S_ + (size_t)((kt + 1) * KT_ + fr))
                                * NE_ + h * D_ + fc;
            const float4* kp = (const float4*)(k + base);
            const float4* vp = (const float4*)(v + base);
#pragma unroll
            for (int i = 0; i < 8; ++i) { sk[i] = kp[i]; sv[i] = vp[i]; }
        }

        // ---- scores: four 16x16 tiles, K-dim = 128 (log2 domain) ----
        v8f sc[4];
#pragma unroll
        for (int kk = 0; kk < 4; ++kk) {
            v8f acc = {};
#pragma unroll
            for (int c = 0; c < 4; ++c) {
                v16bf kf = load_row_frag(&lK[buf][kk * 16 + ln][c * 32], hi);
                acc = __builtin_amdgcn_wmma_f32_16x16x32_bf16(
                          false, qf[c], false, kf, (short)0, acc, false, false);
            }
            sc[kk] = acc;
        }

        // ---- online softmax across 64 keys (base-2 exponentials) ----
        float corr[8];
#pragma unroll
        for (int j = 0; j < 8; ++j) {
            float s0 = sc[0][j], s1 = sc[1][j], s2 = sc[2][j], s3 = sc[3][j];
            float mx   = xmax16(fmaxf(fmaxf(s0, s1), fmaxf(s2, s3)));
            float mnew = fmaxf(m_r[j], mx);
            float cf   = exp2f(m_r[j] - mnew);
            float p0   = exp2f(s0 - mnew);
            float p1   = exp2f(s1 - mnew);
            float p2   = exp2f(s2 - mnew);
            float p3   = exp2f(s3 - mnew);
            l_part[j]  = l_part[j] * cf + ((p0 + p1) + (p2 + p3));
            m_r[j]     = mnew;
            corr[j]    = cf;
            // permuted P columns 4*ln..4*ln+3: contiguous b64 store
            __bf16* pp = &lP[w][j + 8 * hi][4 * ln];
            pp[0] = (__bf16)p0;
            pp[1] = (__bf16)p1;
            pp[2] = (__bf16)p2;
            pp[3] = (__bf16)p3;
        }

        // wave-local: lP stores must land before the A-fragment reload
        asm volatile("s_wait_dscnt 0" ::: "memory");

        v16bf pf0 = load_row_frag(&lP[w][ln][0],  hi);   // perm keys 0..31
        v16bf pf1 = load_row_frag(&lP[w][ln][32], hi);   // perm keys 32..63

        // ---- O rescale: skip (uniform branch) when max didn't move ----
        float cmin = fminf(fminf(fminf(corr[0], corr[1]), fminf(corr[2], corr[3])),
                           fminf(fminf(corr[4], corr[5]), fminf(corr[6], corr[7])));
        if (__any(cmin < 1.0f)) {
#pragma unroll
            for (int dt = 0; dt < 8; ++dt)
#pragma unroll
                for (int j = 0; j < 8; ++j) o_acc[dt][j] *= corr[j];
        }

        // ---- O += P (16x64) x V (64x16 per d-tile) ----
#pragma unroll
        for (int dt = 0; dt < 8; ++dt) {
            v16bf vf0 = load_row_frag(&lVT[buf][dt * 16 + ln][0],  hi);
            v16bf vf1 = load_row_frag(&lVT[buf][dt * 16 + ln][32], hi);
            o_acc[dt] = __builtin_amdgcn_wmma_f32_16x16x32_bf16(
                            false, pf0, false, vf0, (short)0, o_acc[dt], false, false);
            o_acc[dt] = __builtin_amdgcn_wmma_f32_16x16x32_bf16(
                            false, pf1, false, vf1, (short)0, o_acc[dt], false, false);
        }
    }

    // ---- epilogue: reduce lane-partial denominators, normalize, store ----
    float inv_l[8];
#pragma unroll
    for (int j = 0; j < 8; ++j) inv_l[j] = 1.0f / xsum16(l_part[j]);

#pragma unroll
    for (int dt = 0; dt < 8; ++dt) {
#pragma unroll
        for (int j = 0; j < 8; ++j) {
            int row = q0 + j + 8 * hi;
            out[((size_t)b * S_ + row) * NE_ + h * D_ + dt * 16 + ln]
                = o_acc[dt][j] * inv_l[j];
        }
    }
}

extern "C" void kernel_launch(void* const* d_in, const int* in_sizes, int n_in,
                              void* d_out, int out_size, void* d_ws, size_t ws_size,
                              hipStream_t stream) {
    (void)in_sizes; (void)n_in; (void)out_size; (void)d_ws; (void)ws_size;
    const float* q = (const float*)d_in[0];
    const float* k = (const float*)d_in[1];
    const float* v = (const float*)d_in[2];
    float* out = (float*)d_out;
    dim3 grid(S_ / QB_, H_, B_);
    attn_fa_wmma<<<grid, dim3(256), 0, stream>>>(q, k, v, out);
}
